// PointNetFeaturePropagation_16045997818018
// MI455X (gfx1250) — compile-verified, matrix operand order checked by reading, not run on hardware
//
#include <hip/hip_runtime.h>
#include <hip/hip_bf16.h>

typedef __attribute__((ext_vector_type(16))) _Float16 v16h;
typedef __attribute__((ext_vector_type(8)))  float    v8f;

#define BN_EPS 1e-5f
#define INTERP_EPS 1e-8f

// Problem dims (match reference setup_inputs)
#define DB   4
#define DN   16384
#define DS   1024
#define DC1  128
#define DC2  256
#define DCIN 384
#define DH1  256
#define DH2  128

// ---------------------------------------------------------------------------
// Kernel 1: 3-NN + inverse-distance weights. One thread per query point,
// xyz2 of the batch staged in LDS (12 KB).
// ---------------------------------------------------------------------------
__global__ void knn_weights_kernel(const float* __restrict__ xyz1,
                                   const float* __restrict__ xyz2,
                                   float* __restrict__ wout,
                                   int*   __restrict__ iout) {
    __shared__ float s_xyz[DS * 3];
    const int b = blockIdx.y;
    const float* x2 = xyz2 + (size_t)b * DS * 3;
    for (int i = threadIdx.x; i < DS * 3; i += blockDim.x) s_xyz[i] = x2[i];
    __syncthreads();

    const int n = blockIdx.x * blockDim.x + threadIdx.x;
    const size_t p = (size_t)b * DN + n;
    const float px = xyz1[p * 3 + 0];
    const float py = xyz1[p * 3 + 1];
    const float pz = xyz1[p * 3 + 2];

    float d0 = 3.4e38f, d1 = 3.4e38f, d2 = 3.4e38f;
    int   i0 = 0, i1 = 0, i2 = 0;
    for (int s = 0; s < DS; ++s) {
        const float dx = px - s_xyz[3 * s + 0];
        const float dy = py - s_xyz[3 * s + 1];
        const float dz = pz - s_xyz[3 * s + 2];
        const float d  = dx * dx + dy * dy + dz * dz;
        if (d < d0)      { d2 = d1; i2 = i1; d1 = d0; i1 = i0; d0 = d; i0 = s; }
        else if (d < d1) { d2 = d1; i2 = i1; d1 = d;  i1 = s; }
        else if (d < d2) { d2 = d;  i2 = s; }
    }
    const float r0 = 1.0f / (d0 + INTERP_EPS);
    const float r1 = 1.0f / (d1 + INTERP_EPS);
    const float r2 = 1.0f / (d2 + INTERP_EPS);
    const float rs = 1.0f / (r0 + r1 + r2);
    wout[p * 3 + 0] = r0 * rs;  iout[p * 3 + 0] = i0;
    wout[p * 3 + 1] = r1 * rs;  iout[p * 3 + 1] = i1;
    wout[p * 3 + 2] = r2 * rs;  iout[p * 3 + 2] = i2;
}

// ---------------------------------------------------------------------------
// Kernel 2: build new_points = concat(points1, interp(points2)) as f16.
// One block per point, one thread per channel (384 threads = 12 waves).
// ---------------------------------------------------------------------------
__global__ void build_new_points_kernel(const float* __restrict__ points1,
                                        const float* __restrict__ points2,
                                        const float* __restrict__ wgt,
                                        const int*   __restrict__ idx,
                                        _Float16*    __restrict__ np) {
    const size_t p = blockIdx.x;          // 0 .. B*N-1
    const int    c = threadIdx.x;         // 0 .. 383
    const int    b = (int)(p / DN);

    __shared__ float sw[3];
    __shared__ int   si[3];
    if (c < 3) { sw[c] = wgt[p * 3 + c]; si[c] = idx[p * 3 + c]; }
    __syncthreads();

    float v;
    if (c < DC1) {
        v = points1[p * DC1 + c];
    } else {
        const int cc = c - DC1;
        const float* base = points2 + (size_t)b * DS * DC2;
        v = sw[0] * base[(size_t)si[0] * DC2 + cc]
          + sw[1] * base[(size_t)si[1] * DC2 + cc]
          + sw[2] * base[(size_t)si[2] * DC2 + cc];
    }
    np[p * DCIN + c] = (_Float16)v;
}

// ---------------------------------------------------------------------------
// Kernel 3: convert weights to f16, fold BN into (scale, shift).
// ---------------------------------------------------------------------------
__global__ void prep_params_kernel(const float* __restrict__ w1, const float* __restrict__ w2,
                                   const float* __restrict__ g1, const float* __restrict__ b1,
                                   const float* __restrict__ m1, const float* __restrict__ v1,
                                   const float* __restrict__ g2, const float* __restrict__ b2,
                                   const float* __restrict__ m2, const float* __restrict__ v2,
                                   _Float16* __restrict__ w1h, _Float16* __restrict__ w2h,
                                   float* __restrict__ s1, float* __restrict__ t1,
                                   float* __restrict__ s2, float* __restrict__ t2) {
    const int tid    = blockIdx.x * blockDim.x + threadIdx.x;
    const int stride = gridDim.x * blockDim.x;
    for (int j = tid; j < DH1 * DCIN; j += stride) w1h[j] = (_Float16)w1[j];
    for (int j = tid; j < DH2 * DH1;  j += stride) w2h[j] = (_Float16)w2[j];
    if (tid < DH1) {
        const float s = g1[tid] / sqrtf(v1[tid] + BN_EPS);
        s1[tid] = s; t1[tid] = b1[tid] - m1[tid] * s;
    }
    if (tid < DH2) {
        const float s = g2[tid] / sqrtf(v2[tid] + BN_EPS);
        s2[tid] = s; t2[tid] = b2[tid] - m2[tid] * s;
    }
}

// ---------------------------------------------------------------------------
// Kernel 4: WMMA GEMM + fused BN + ReLU.
//   out[m, n] = relu( (sum_k A[m,k] * W[n,k]) * scale[n] + shift[n] )
// A: M x K f16 row-major. W: N x K f16 row-major (W row n == ISA B-fragment
// column n, so no transpose is needed). Each wave owns a 16(M) x 64(N) output
// strip: one A fragment is reused by 4 WMMAs per 32-deep K step, giving
// 4 v_wmma per 10 global_load_b128 instead of 1 per 4.
// ---------------------------------------------------------------------------
union FragA { v16h h; unsigned u[8]; };

template <bool OUT_F32>
__global__ void wmma_gemm_bn_relu(const _Float16* __restrict__ A,
                                  const _Float16* __restrict__ W,
                                  const float* __restrict__ scale,
                                  const float* __restrict__ shift,
                                  void* __restrict__ outp,
                                  int M, int N, int K, int superN /* = N/64 */) {
    const int wave = threadIdx.x >> 5;
    const int lane = threadIdx.x & 31;
    const int half = lane >> 4;     // which 16-lane half of the wave
    const int lm   = lane & 15;     // row (A) / col (B,C) within tile

    const int tile   = blockIdx.x * 8 + wave;
    const int tile_m = tile / superN;
    const int tile_n = (tile - tile_m * superN) * 4;   // in 16-column units

    const _Float16* __restrict__ arow = A + (size_t)(tile_m * 16 + lm) * K;
    const _Float16* __restrict__ brow = W + (size_t)(tile_n * 16 + lm) * K;
    const size_t bstride = (size_t)16 * K;             // next 16-column tile

    v8f acc[4] = {v8f{}, v8f{}, v8f{}, v8f{}};
    for (int k0 = 0; k0 < K; k0 += 32) {
        __builtin_prefetch(arow + k0 + 32, 0, 0);      // global_prefetch_b8
        FragA a, b[4];
#pragma unroll
        for (int v = 0; v < 8; ++v) {
            // ISA 16-bit fragment layout: VGPR v holds K pair
            //   k = (v>=4 ? 16 : 0) + (v&3)*2 + half*8  (and k+1)
            const int k = k0 + ((v & 4) << 2) + ((v & 3) << 1) + (half << 3);
            a.u[v] = *(const unsigned*)(arow + k);
#pragma unroll
            for (int nn = 0; nn < 4; ++nn)
                b[nn].u[v] = *(const unsigned*)(brow + (size_t)nn * bstride + k);
        }
#pragma unroll
        for (int nn = 0; nn < 4; ++nn)
            acc[nn] = __builtin_amdgcn_wmma_f32_16x16x32_f16(
                /*neg_a=*/false, a.h, /*neg_b=*/false, b[nn].h,
                /*c_mod=*/(short)0, acc[nn], /*reuse_a=*/false, /*reuse_b=*/false);
    }

    // Epilogue: BN params depend only on the output column.
#pragma unroll
    for (int nn = 0; nn < 4; ++nn) {
        const int col = (tile_n + nn) * 16 + lm;
        const float s = scale[col];
        const float t = shift[col];
#pragma unroll
        for (int r = 0; r < 8; ++r) {
            const int row = tile_m * 16 + r + half * 8;  // C/D layout: VGPR r -> row
            float v = fmaxf(acc[nn][r] * s + t, 0.0f);
            if (OUT_F32) ((float*)outp)[(size_t)row * N + col] = v;
            else         ((_Float16*)outp)[(size_t)row * N + col] = (_Float16)v;
        }
    }
}

// ---------------------------------------------------------------------------
// Launcher
// ---------------------------------------------------------------------------
extern "C" void kernel_launch(void* const* d_in, const int* in_sizes, int n_in,
                              void* d_out, int out_size, void* d_ws, size_t ws_size,
                              hipStream_t stream) {
    const float* xyz1    = (const float*)d_in[0];
    const float* xyz2    = (const float*)d_in[1];
    const float* points1 = (const float*)d_in[2];
    const float* points2 = (const float*)d_in[3];
    const float* w1 = (const float*)d_in[4];
    const float* g1 = (const float*)d_in[5];
    const float* b1 = (const float*)d_in[6];
    const float* m1 = (const float*)d_in[7];
    const float* v1 = (const float*)d_in[8];
    const float* w2 = (const float*)d_in[9];
    const float* g2 = (const float*)d_in[10];
    const float* b2 = (const float*)d_in[11];
    const float* m2 = (const float*)d_in[12];
    const float* v2 = (const float*)d_in[13];

    const size_t P = (size_t)DB * DN;   // 65536 points

    // Workspace layout (256B aligned slices)
    char* ws = (char*)d_ws;
    size_t off = 0;
    auto take = [&](size_t bytes) { char* p = ws + off; off += (bytes + 255) & ~(size_t)255; return p; };
    float*    knn_w = (float*)   take(P * 3 * sizeof(float));
    int*      knn_i = (int*)     take(P * 3 * sizeof(int));
    _Float16* np    = (_Float16*)take(P * DCIN * sizeof(_Float16)); // 48 MB
    _Float16* hbuf  = (_Float16*)take(P * DH1  * sizeof(_Float16)); // 32 MB
    _Float16* w1h   = (_Float16*)take((size_t)DH1 * DCIN * sizeof(_Float16));
    _Float16* w2h   = (_Float16*)take((size_t)DH2 * DH1  * sizeof(_Float16));
    float*    s1    = (float*)   take(DH1 * sizeof(float));
    float*    t1    = (float*)   take(DH1 * sizeof(float));
    float*    s2    = (float*)   take(DH2 * sizeof(float));
    float*    t2    = (float*)   take(DH2 * sizeof(float));
    (void)ws_size; (void)in_sizes; (void)n_in; (void)out_size;

    // 1) 3-NN + weights
    knn_weights_kernel<<<dim3(DN / 256, DB), 256, 0, stream>>>(xyz1, xyz2, knn_w, knn_i);

    // 2) interpolate + concat -> f16 activations
    build_new_points_kernel<<<(unsigned)P, DCIN, 0, stream>>>(points1, points2, knn_w, knn_i, np);

    // 3) weights -> f16, BN folding (independent of 1/2)
    prep_params_kernel<<<128, 256, 0, stream>>>(w1, w2, g1, b1, m1, v1, g2, b2, m2, v2,
                                                w1h, w2h, s1, t1, s2, t2);

    // 4) GEMM1: (P x 384) x (384 x 256) + BN + ReLU -> h (f16)
    {
        const int superN = DH1 / 64;                       // 4
        const int waves  = (int)(P / 16) * superN;         // 16384
        wmma_gemm_bn_relu<false><<<waves / 8, 256, 0, stream>>>(
            np, w1h, s1, t1, hbuf, (int)P, DH1, DCIN, superN);
    }

    // 5) GEMM2: (P x 256) x (256 x 128) + BN + ReLU -> out (f32)
    {
        const int superN = DH2 / 64;                       // 2
        const int waves  = (int)(P / 16) * superN;         // 8192
        wmma_gemm_bn_relu<true><<<waves / 8, 256, 0, stream>>>(
            hbuf, w2h, s2, t2, d_out, (int)P, DH2, DH1, superN);
    }
}
